// DimRnn_47742856463178
// MI455X (gfx1250) — matrix operand order repaired and checked by reading, not compiled, not run on hardware
//
#include <hip/hip_runtime.h>
#include <hip/hip_bf16.h>
#include <math.h>

#define B_ 16
#define T_ 512
#define E_ 2048
#define H_ 1024
#define D_ 128

#define NWG 32            // persistent workgroups for the recurrence
#define ROWS_PER_WG 32    // H_ / NWG
#define AROW 2056         // padded LDS row (elements): 4112B stride, bank-friendly

typedef __attribute__((ext_vector_type(16))) __bf16 v16bf;
typedef __attribute__((ext_vector_type(8)))  __bf16 v8bf;
typedef __attribute__((ext_vector_type(8)))  float  v8f;

union V16 { v16bf v; v8bf h[2]; unsigned int u[8]; };
union V8  { v8f v; float f[8]; };

// --- CDNA5 async global->LDS path (guarded: toolchains differ) --------------
#if defined(__HIP_DEVICE_COMPILE__) && \
    __has_builtin(__builtin_amdgcn_global_load_async_to_lds_b128) && \
    __has_builtin(__builtin_amdgcn_s_wait_asynccnt)
#define HAVE_ASYNC_LDS 1
#else
#define HAVE_ASYNC_LDS 0
#endif

#define AS1 __attribute__((address_space(1)))
#define AS3 __attribute__((address_space(3)))

#if HAVE_ASYNC_LDS
// builtin prototype is void(v4i AS1*, v4i AS3*, Imm offset, Imm cpol)
typedef int v4i __attribute__((vector_size(16)));
__device__ __forceinline__ void async_b128(const void* g, void* l) {
  // LDS byte offset is the low 32 bits of the generic shared address (ISA 10.2)
  __builtin_amdgcn_global_load_async_to_lds_b128(
      (AS1 v4i*)(unsigned long long)g,
      (AS3 v4i*)(unsigned int)(unsigned long long)l, 0, 0);
}
#endif

__device__ __forceinline__ v8f wmma_bf16(v16bf a, v16bf b, v8f c) {
  // (neg_a, A, neg_b, B, c_mod, C, reuse_a, reuse_b)
  return __builtin_amdgcn_wmma_f32_16x16x32_bf16(false, a, false, b, (short)0, c,
                                                 false, false);
}

// ---------------------------------------------------------------------------
// fp32 -> (hi, lo) bf16 split:  v ~= hi + lo,  keeps ~fp32 accuracy in bf16x3
// ---------------------------------------------------------------------------
__global__ void k_split_bf16(const float* __restrict__ in, __bf16* __restrict__ hi,
                             __bf16* __restrict__ lo, int n) {
  int i = blockIdx.x * blockDim.x + threadIdx.x;
  if (i < n) {
    float v = in[i];
    __bf16 h = (__bf16)v;
    hi[i] = h;
    lo[i] = (__bf16)(v - (float)h);
  }
}

// ---------------------------------------------------------------------------
// Phase 1: xW[m, n] = sum_k x[m,k] * W_ih[n,k] + b_ih[n] + b_hh[n]
//   M=8192, N=1024, K=2048.  Block = one 16-row x strip: the full 16x2048
//   hi+lo A-panel (128 KB) is staged into LDS once with async-to-LDS loads,
//   then 8 waves x 8 tiles read A from LDS and stream B (W_ih) from L2.
// ---------------------------------------------------------------------------
__global__ void __launch_bounds__(256) k_xw_gemm(
    const __bf16* __restrict__ xhi, const __bf16* __restrict__ xlo,
    const __bf16* __restrict__ whi, const __bf16* __restrict__ wlo,
    const float* __restrict__ b_ih, const float* __restrict__ b_hh,
    float* __restrict__ xw) {
  __shared__ __align__(64) __bf16 a_hi[16 * AROW];   // ~64.3 KB
  __shared__ __align__(64) __bf16 a_lo[16 * AROW];   // ~64.3 KB

  const int tid  = threadIdx.x;
  const int lane = tid & 31;
  const int wave = tid >> 5;
  const int mt   = blockIdx.x;            // 512 row strips
  const int ml   = lane & 15;             // A row / B col within tile
  const int hf   = lane >> 4;             // lane half selects K sub-range

  // stage the A panel (16 rows x 2048, hi+lo) into LDS
  for (int i = tid; i < 16 * (E_ / 8); i += 256) {   // 4096 16B chunks per half
    const int r = i / (E_ / 8);
    const int c = i % (E_ / 8);
    const v8bf* sh = (const v8bf*)(xhi + (size_t)(mt * 16 + r) * E_) + c;
    const v8bf* sl = (const v8bf*)(xlo + (size_t)(mt * 16 + r) * E_) + c;
    v8bf* dh = (v8bf*)(a_hi + r * AROW) + c;
    v8bf* dl = (v8bf*)(a_lo + r * AROW) + c;
#if HAVE_ASYNC_LDS
    async_b128(sh, dh);
    async_b128(sl, dl);
#else
    *dh = *sh; *dl = *sl;
#endif
  }
#if HAVE_ASYNC_LDS
  __builtin_amdgcn_s_wait_asynccnt(0);
#endif
  __syncthreads();

  const __bf16* la_hi = a_hi + ml * AROW;
  const __bf16* la_lo = a_lo + ml * AROW;

  for (int ntl = 0; ntl < 8; ++ntl) {
    const int nt = wave * 8 + ntl;        // 64 column tiles, 8 per wave
    const __bf16* brow_hi = whi + (size_t)(nt * 16 + ml) * E_;
    const __bf16* brow_lo = wlo + (size_t)(nt * 16 + ml) * E_;

    v8f acc = {};
    for (int kt = 0; kt < E_ / 32; ++kt) {
      // A fragment (16x32) from LDS: K = {hf*8..+8} and {16+hf*8..+8}
      const int k0 = kt * 32 + hf * 8;
      V16 ah, al, bh, bl;
      ah.h[0] = *(const v8bf*)(la_hi + k0);
      ah.h[1] = *(const v8bf*)(la_hi + k0 + 16);
      al.h[0] = *(const v8bf*)(la_lo + k0);
      al.h[1] = *(const v8bf*)(la_lo + k0 + 16);
      // B fragment (32x16) from global: K = {hf*8..+8} and {16+hf*8..+8} of column ml
      const int kb = kt * 32 + hf * 8;
      bh.h[0] = *(const v8bf*)(brow_hi + kb);
      bh.h[1] = *(const v8bf*)(brow_hi + kb + 16);
      bl.h[0] = *(const v8bf*)(brow_lo + kb);
      bl.h[1] = *(const v8bf*)(brow_lo + kb + 16);
      acc = wmma_bf16(ah.v, bh.v, acc);   // hi*hi
      acc = wmma_bf16(ah.v, bl.v, acc);   // hi*lo
      acc = wmma_bf16(al.v, bh.v, acc);   // lo*hi
    }

    const int ncol = nt * 16 + ml;        // D: n = lane & 15
    const float bias = b_ih[ncol] + b_hh[ncol];
    V8 a; a.v = acc;
    #pragma unroll
    for (int r = 0; r < 8; ++r) {         // D: m = r + 8*(lane>>4)
      const int mrow = mt * 16 + hf * 8 + r;
      xw[(size_t)mrow * H_ + ncol] = a.f[r] + bias;
    }
  }
}

// ---------------------------------------------------------------------------
// Global split-barrier over NWG persistent workgroups (monotonic counter).
// ---------------------------------------------------------------------------
__device__ __forceinline__ void grid_barrier(unsigned int* ctr, unsigned int target,
                                             int tid) {
  __syncthreads();
  if (tid == 0) {
    __hip_atomic_fetch_add(ctr, 1u, __ATOMIC_RELEASE, __HIP_MEMORY_SCOPE_AGENT);
    while (__hip_atomic_load(ctr, __ATOMIC_RELAXED, __HIP_MEMORY_SCOPE_AGENT) <
           target) {
      __builtin_amdgcn_s_sleep(2);
    }
  }
  __syncthreads();
  __threadfence();  // acquire: invalidate stale cached h lines
}

// ---------------------------------------------------------------------------
// Phase 2: sequential recurrence. Each WG keeps its 32x1024 slice of W_hh in
// LDS as bf16 hi+lo (128 KB - needs CDNA5's 320 KB LDS), staged via async
// loads. Per step: WMMA matvec (h in column 0 of B), cross-wave LDS reduce,
// tanh, publish 32 elements, one global barrier (h double-buffered).
// ---------------------------------------------------------------------------
__global__ void __launch_bounds__(256, 1) k_rnn(
    const __bf16* __restrict__ whh_hi, const __bf16* __restrict__ whh_lo,
    const float* __restrict__ xw, const int* __restrict__ lengths,
    float* __restrict__ hbuf /*[2][H_]*/, float* __restrict__ hs /*[B_][H_]*/,
    unsigned int* __restrict__ ctr) {
  __shared__ __align__(64) __bf16 ws_hi[ROWS_PER_WG * H_];   // 64 KB
  __shared__ __align__(64) __bf16 ws_lo[ROWS_PER_WG * H_];   // 64 KB
  __shared__ __align__(64) __bf16 hv_hi[H_];                 // 2 KB
  __shared__ __align__(64) __bf16 hv_lo[H_];                 // 2 KB
  __shared__ float partial[8][16];

  const int tid  = threadIdx.x;
  const int lane = tid & 31;
  const int wave = tid >> 5;
  const int wg   = blockIdx.x;
  const int rowbase = wg * ROWS_PER_WG;
  const int ml = lane & 15;
  const int hf = lane >> 4;

  // stage this WG's W_hh slice into LDS (async global->LDS when available)
  {
    const v8bf* sh = (const v8bf*)(whh_hi + (size_t)rowbase * H_);
    const v8bf* sl = (const v8bf*)(whh_lo + (size_t)rowbase * H_);
    v8bf* dh = (v8bf*)ws_hi;
    v8bf* dl = (v8bf*)ws_lo;
    const int nchunk = ROWS_PER_WG * H_ / 8;  // 4096
    for (int i = tid; i < nchunk; i += 256) {
#if HAVE_ASYNC_LDS
      async_b128(sh + i, dh + i);
      async_b128(sl + i, dl + i);
#else
      dh[i] = sh[i]; dl[i] = sl[i];
#endif
    }
#if HAVE_ASYNC_LDS
    __builtin_amdgcn_s_wait_asynccnt(0);
#endif
  }

  // h0 = 0 in buffer 0
  if (tid < ROWS_PER_WG) hbuf[rowbase + tid] = 0.0f;
  __threadfence();
  unsigned int nbar = 1;
  grid_barrier(ctr, nbar * NWG, tid);

  const int rt = wave & 1;        // which 16-row tile of this WG's 32 rows
  const int kg = wave >> 1;       // K-group: 8 k-tiles per wave
  const __bf16* wrow_hi = ws_hi + (size_t)(rt * 16 + ml) * H_;
  const __bf16* wrow_lo = ws_lo + (size_t)(rt * 16 + ml) * H_;

  int s = 0;
  for (int b = 0; b < B_; ++b) {
    const int len = lengths[b];
    for (int t = 0; t < len; ++t) {
      const float* hcur = hbuf + (size_t)(s & 1) * H_;
      float* hnext      = hbuf + (size_t)((s + 1) & 1) * H_;

      // convert current hidden state to bf16 hi/lo in LDS (4 values/thread)
      {
        const int i = tid * 4;
        const float4 v = *(const float4*)(hcur + i);
        const float vv[4] = {v.x, v.y, v.z, v.w};
        #pragma unroll
        for (int c = 0; c < 4; ++c) {
          __bf16 h = (__bf16)vv[c];
          hv_hi[i + c] = h;
          hv_lo[i + c] = (__bf16)(vv[c] - (float)h);
        }
      }
      __syncthreads();

      // y[rt*16 + m] partial over k-tiles [kg*8, kg*8+8)
      v8f acc = {};
      for (int kk = 0; kk < 8; ++kk) {
        const int kt = kg * 8 + kk;
        const int k0 = kt * 32 + hf * 8;
        V16 ah, al, bh, bl;
        ah.h[0] = *(const v8bf*)(wrow_hi + k0);
        ah.h[1] = *(const v8bf*)(wrow_hi + k0 + 16);
        al.h[0] = *(const v8bf*)(wrow_lo + k0);
        al.h[1] = *(const v8bf*)(wrow_lo + k0 + 16);
        if (ml == 0) {  // h occupies only column n==0 of B
          bh.h[0] = *(const v8bf*)(hv_hi + kt * 32 + hf * 8);
          bh.h[1] = *(const v8bf*)(hv_hi + kt * 32 + hf * 8 + 16);
          bl.h[0] = *(const v8bf*)(hv_lo + kt * 32 + hf * 8);
          bl.h[1] = *(const v8bf*)(hv_lo + kt * 32 + hf * 8 + 16);
        } else {
          #pragma unroll
          for (int q = 0; q < 8; ++q) { bh.u[q] = 0u; bl.u[q] = 0u; }
        }
        acc = wmma_bf16(ah.v, bh.v, acc);
        acc = wmma_bf16(ah.v, bl.v, acc);
        acc = wmma_bf16(al.v, bh.v, acc);
      }

      // column 0 of D lives in lanes 0 (m=0..7) and 16 (m=8..15)
      if (ml == 0) {
        V8 a; a.v = acc;
        const int mb = hf * 8;
        #pragma unroll
        for (int r = 0; r < 8; ++r) partial[wave][mb + r] = a.f[r];
      }
      __syncthreads();

      if (tid < ROWS_PER_WG) {
        const int rt2 = tid >> 4, m = tid & 15;
        const float y = partial[rt2][m] + partial[rt2 + 2][m] +
                        partial[rt2 + 4][m] + partial[rt2 + 6][m];
        const float xwv = xw[(size_t)(b * T_ + t) * H_ + rowbase + tid];
        const float hn = tanhf(xwv + y);
        hnext[rowbase + tid] = hn;
        if (t == len - 1) hs[(size_t)b * H_ + rowbase + tid] = hn;
        __threadfence();  // publish before barrier arrive
      }
      ++nbar;
      grid_barrier(ctr, nbar * NWG, tid);
      ++s;
    }
  }
}

// ---------------------------------------------------------------------------
// Phase 3: out[b,d] = hs[b,:] . W_l1[d,:] + b_l1[d]    (tiny: 16x128)
// ---------------------------------------------------------------------------
__global__ void k_out(const float* __restrict__ hs, const float* __restrict__ wl1,
                      const float* __restrict__ bl1, float* __restrict__ out) {
  const int j = blockIdx.x * blockDim.x + threadIdx.x;
  if (j >= B_ * D_) return;
  const int b = j / D_, d = j % D_;
  const float* hv = hs + (size_t)b * H_;
  const float* wv = wl1 + (size_t)d * H_;
  float acc = bl1[d];
  for (int k = 0; k < H_; ++k) acc += hv[k] * wv[k];
  out[j] = acc;
}

// ---------------------------------------------------------------------------
extern "C" void kernel_launch(void* const* d_in, const int* in_sizes, int n_in,
                              void* d_out, int out_size, void* d_ws, size_t ws_size,
                              hipStream_t stream) {
  (void)in_sizes; (void)n_in; (void)out_size; (void)ws_size;
  const float* x   = (const float*)d_in[0];
  const int*   len = (const int*)d_in[1];
  const float* Wih = (const float*)d_in[2];
  const float* Whh = (const float*)d_in[3];
  const float* bih = (const float*)d_in[4];
  const float* bhh = (const float*)d_in[5];
  const float* Wl1 = (const float*)d_in[6];
  const float* bl1 = (const float*)d_in[7];
  float* out = (float*)d_out;

  // workspace carve-up (256B aligned)
  char* ws = (char*)d_ws;
  size_t off = 0;
  auto carve = [&](size_t bytes) -> char* {
    char* p = ws + off;
    off = (off + bytes + 255) & ~(size_t)255;
    return p;
  };
  const size_t n_x   = (size_t)B_ * T_ * E_;  // 16,777,216
  const size_t n_wih = (size_t)H_ * E_;       //  2,097,152
  const size_t n_whh = (size_t)H_ * H_;       //  1,048,576

  float*  xw     = (float*)carve((size_t)B_ * T_ * H_ * 4);
  __bf16* x_hi   = (__bf16*)carve(n_x * 2);
  __bf16* x_lo   = (__bf16*)carve(n_x * 2);
  __bf16* wih_hi = (__bf16*)carve(n_wih * 2);
  __bf16* wih_lo = (__bf16*)carve(n_wih * 2);
  __bf16* whh_hi = (__bf16*)carve(n_whh * 2);
  __bf16* whh_lo = (__bf16*)carve(n_whh * 2);
  float*  hbuf   = (float*)carve(2 * H_ * 4);
  float*  hs     = (float*)carve((size_t)B_ * H_ * 4);
  unsigned int* ctr = (unsigned int*)carve(256);

  (void)hipMemsetAsync(ctr, 0, 256, stream);  // deterministic barrier reset

  k_split_bf16<<<(int)(n_x   / 256), 256, 0, stream>>>(x,   x_hi,   x_lo,   (int)n_x);
  k_split_bf16<<<(int)(n_wih / 256), 256, 0, stream>>>(Wih, wih_hi, wih_lo, (int)n_wih);
  k_split_bf16<<<(int)(n_whh / 256), 256, 0, stream>>>(Whh, whh_hi, whh_lo, (int)n_whh);

  // one block per 16-row x strip; A panel staged in LDS, 64 tiles per block
  k_xw_gemm<<<512, 256, 0, stream>>>(x_hi, x_lo, wih_hi, wih_lo, bih, bhh, xw);

  k_rnn<<<NWG, 256, 0, stream>>>(whh_hi, whh_lo, xw, len, hbuf, hs, ctr);

  k_out<<<8, 256, 0, stream>>>(hs, Wl1, bl1, out);
}
